// SAVSS_2D_27865747816471
// MI455X (gfx1250) — matrix operand order repaired and checked by reading, not compiled, not run on hardware
//
#include <hip/hip_runtime.h>
#include <hip/hip_bf16.h>

// ---------------------------------------------------------------------------
// SAVSS_2D forward for MI455X (gfx1250, wave32).
//   GEMMs      : v_wmma_f32_16x16x32_f16, 8 waves/block, 128xNT*16 block tile,
//                weights staged in LDS via global_load_async_to_lds_b128
//                (ASYNCcnt-tracked), A fragment register-reused across NT tiles.
//   Scan       : 1 lane per (seq, channel), 16-state in VGPRs, 192 waves,
//                exp via v_exp_f32, scatter via global_atomic_add_f32,
//                global_prefetch_b8 for next-step gathered rows.
//   Conv stack : negligible FLOPs, plain VALU kernels.
// ---------------------------------------------------------------------------

#define D_MODEL 96
#define D_STATE 16
#define E_DIM   192
#define MID_DIM 12
#define R_DIM   6
#define BATCH   8
#define HH      64
#define WW      64
#define L_SEQ   (HH * WW)            // 4096
#define BL      (BATCH * L_SEQ)      // 32768
#define XPROJ_OUT (R_DIM + 2 * D_STATE)  // 38

typedef __attribute__((ext_vector_type(16))) _Float16 v16h;
typedef __attribute__((ext_vector_type(8)))  _Float16 v8h;
typedef __attribute__((ext_vector_type(8)))  float    v8f;
typedef __attribute__((ext_vector_type(4)))  int      v4i;

#ifndef __has_builtin
#define __has_builtin(x) 0
#endif
#if __has_builtin(__builtin_amdgcn_global_load_async_to_lds_b128) && \
    __has_builtin(__builtin_amdgcn_s_wait_asynccnt)
#define USE_ASYNC_LDS 1
#else
#define USE_ASYNC_LDS 0
#endif

// ---------------------------------------------------------------------------
// WMMA fragment helpers (layouts per CDNA5 ISA §7.12.2, wave32)
// ---------------------------------------------------------------------------

// A fragment: 16x32 f16 (row-major A, lda in elements).
// lanes 0-15:  row = lane,    K = k0+{0..7, 16..23}
// lanes 16-31: row = lane-16, K = k0+{8..15, 24..31}
__device__ __forceinline__ v16h load_a_frag(const _Float16* __restrict__ A,
                                            int lda, int row0, int k0) {
    int lane = threadIdx.x & 31;
    int row  = row0 + (lane & 15);
    int koff = (lane < 16) ? 0 : 8;
    const _Float16* p = A + (size_t)row * lda + k0 + koff;
    v8h lo = *(const v8h*)(p);        // K = koff..koff+7
    v8h hi = *(const v8h*)(p + 16);   // K = koff+16..koff+23
    v16h r;
#pragma unroll
    for (int i = 0; i < 8; ++i) { r[i] = lo[i]; r[8 + i] = hi[i]; }
    return r;
}

// B fragment from an LDS-resident weight strip (rows x K halves, row-major).
// B(k,n) = Wstrip[n][k]. lane l: n = n0loc + (l&15);
// lanes 0-15 hold K = k0+{0..7, 16..23}, lanes 16-31 hold K = k0+{8..15, 24..31}.
__device__ __forceinline__ v16h load_b_frag_lds(const _Float16* lb,
                                                int ldw, int n0loc, int k0) {
    int lane = threadIdx.x & 31;
    int n    = n0loc + (lane & 15);
    int koff = (lane < 16) ? 0 : 8;
    const _Float16* p = lb + (size_t)n * ldw + k0 + koff;
    v8h lo = *(const v8h*)(p);
    v8h hi = *(const v8h*)(p + 16);
    v16h r;
#pragma unroll
    for (int i = 0; i < 8; ++i) { r[i] = lo[i]; r[8 + i] = hi[i]; }
    return r;
}

// C/D: v8f, lane l holds col = l&15, rows row0 + (l>>4)*8 + {0..7}
__device__ __forceinline__ void store_c_frag(float* __restrict__ C, int ldc,
                                             int row0, int col0, v8f acc) {
    int lane  = threadIdx.x & 31;
    int col   = col0 + (lane & 15);
    int rbase = row0 + ((lane >> 4) << 3);
#pragma unroll
    for (int i = 0; i < 8; ++i)
        C[(size_t)(rbase + i) * ldc + col] = acc[i];
}

// ---------------------------------------------------------------------------
// Tiled WMMA GEMM: C = A(MxK, f16) * W(NxK, f16)^T, f32 out.
// Block = 256 threads (8 waves). Block tile = 128 rows x NT*16 cols.
// W strip (NT*16 rows x K) staged in LDS once, shared by all 8 waves;
// A fragment is loaded once per k-step and reused across NT WMMAs.
// Requires M % 128 == 0, N % (NT*16) == 0, K % 32 == 0.
// ---------------------------------------------------------------------------
template <int NT>
__global__ void wmma_gemm_nt(const _Float16* __restrict__ A,
                             const _Float16* __restrict__ W,
                             float* __restrict__ C,
                             int M, int N, int K) {
    extern __shared__ _Float16 lb[];          // NT*16 x K halves
    const int n0     = blockIdx.y * (NT * 16);
    const int mBase  = blockIdx.x * 128;

    // ---- stage weight strip into LDS (16B chunks) -----------------------
    {
        const int chunks = (NT * 16 * K * 2) / 16;
        const char* gsrc = (const char*)(W + (size_t)n0 * K);
        char*       ldst = (char*)lb;
        for (int c = threadIdx.x; c < chunks; c += blockDim.x) {
#if USE_ASYNC_LDS
            __builtin_amdgcn_global_load_async_to_lds_b128(
                (__attribute__((address_space(1))) v4i*)(unsigned long long)
                    (gsrc + (size_t)c * 16),
                (__attribute__((address_space(3))) v4i*)(unsigned int)
                    (unsigned long long)(ldst + (size_t)c * 16),
                0, 0);
#else
            ((v4i*)ldst)[c] = ((const v4i*)gsrc)[c];
#endif
        }
#if USE_ASYNC_LDS
        __builtin_amdgcn_s_wait_asynccnt(0);
#endif
        __syncthreads();
    }

    const int tm = mBase + ((threadIdx.x >> 5) << 4);   // this wave's M tile
    (void)M;

    v8f acc[NT];
#pragma unroll
    for (int j = 0; j < NT; ++j) acc[j] = (v8f){};

    for (int k0 = 0; k0 < K; k0 += 32) {
        v16h a = load_a_frag(A, K, tm, k0);             // loaded once, reused NT x
#pragma unroll
        for (int j = 0; j < NT; ++j) {
            v16h b = load_b_frag_lds(lb, K, j * 16, k0);
            acc[j] = __builtin_amdgcn_wmma_f32_16x16x32_f16(
                /*neg_a=*/false, a, /*neg_b=*/false, b,
                /*c_mod=*/(short)0, acc[j], /*reuse_a=*/false, /*reuse_b=*/false);
        }
    }
#pragma unroll
    for (int j = 0; j < NT; ++j)
        store_c_frag(C, N, tm, n0 + j * 16, acc[j]);
}

// ---------------------------------------------------------------------------
// Utility kernels
// ---------------------------------------------------------------------------

__global__ void f32_to_f16_kernel(const float* __restrict__ in,
                                  _Float16* __restrict__ out, int n) {
    int i = blockIdx.x * blockDim.x + threadIdx.x;
    if (i < n) out[i] = (_Float16)in[i];
}

__global__ void zero_f32_kernel(float* __restrict__ p, int n) {
    int i = blockIdx.x * blockDim.x + threadIdx.x;
    if (i < n) p[i] = 0.0f;
}

// ---------------------------------------------------------------------------
// Conv stack (tiny): pw1 -> dw3x3 -> pw2 -> silu
// ---------------------------------------------------------------------------

// h1[b][m][p] = pw1_b[m] + sum_e xz[b,p,e] * pw1_w[m,e]   (xi = xz[:, :192])
__global__ void pw1_kernel(const float* __restrict__ xz,
                           const float* __restrict__ w,
                           const float* __restrict__ bias,
                           float* __restrict__ h1) {
    int i = blockIdx.x * blockDim.x + threadIdx.x;
    if (i >= BATCH * MID_DIM * L_SEQ) return;
    int p = i % L_SEQ;
    int m = (i / L_SEQ) % MID_DIM;
    int b = i / (L_SEQ * MID_DIM);
    const float* xrow = xz + ((size_t)b * L_SEQ + p) * (2 * E_DIM);
    const float* wrow = w + m * E_DIM;
    float acc = bias[m];
#pragma unroll 4
    for (int e = 0; e < E_DIM; ++e) acc += xrow[e] * wrow[e];
    h1[i] = acc;
}

// Fused dw3x3 (zero pad) + pw2 + silu. One block (192 thr) per (b, pixel).
__global__ void dwpw2_kernel(const float* __restrict__ h1,
                             const float* __restrict__ dw_w,
                             const float* __restrict__ pw2_w,
                             float* __restrict__ x_conv) {
    int bp = blockIdx.x;          // 0..BL-1
    int b  = bp >> 12;
    int p  = bp & (L_SEQ - 1);
    int ph = p >> 6, pw = p & 63;

    __shared__ float dwv[MID_DIM];
    int e = threadIdx.x;
    if (e < MID_DIM) {
        const float* hc = h1 + ((size_t)b * MID_DIM + e) * L_SEQ;
        const float* wk = dw_w + e * 9;
        float acc = 0.0f;
#pragma unroll
        for (int dy = -1; dy <= 1; ++dy)
#pragma unroll
            for (int dx = -1; dx <= 1; ++dx) {
                int y = ph + dy, x = pw + dx;
                if (y >= 0 && y < HH && x >= 0 && x < WW)
                    acc += hc[y * WW + x] * wk[(dy + 1) * 3 + (dx + 1)];
            }
        dwv[e] = acc;
    }
    __syncthreads();
    float acc = 0.0f;
    const float* w2 = pw2_w + e * MID_DIM;
#pragma unroll
    for (int m = 0; m < MID_DIM; ++m) acc += dwv[m] * w2[m];
    float sg = 1.0f / (1.0f + __expf(-acc));         // silu
    x_conv[((size_t)b * L_SEQ + p) * E_DIM + e] = acc * sg;
}

// ---------------------------------------------------------------------------
// x_dbl = x_conv @ x_proj_w^T  (38 outputs/row); dt = x_dbl[:, :6] @ dt_proj_w^T
// ---------------------------------------------------------------------------

__global__ void xdbl_kernel(const float* __restrict__ x_conv,
                            const float* __restrict__ x_proj_w,
                            float* __restrict__ x_dbl) {
    int i = blockIdx.x * blockDim.x + threadIdx.x;
    if (i >= BL * XPROJ_OUT) return;
    int j   = i % XPROJ_OUT;
    int row = i / XPROJ_OUT;
    const float* xr = x_conv + (size_t)row * E_DIM;
    const float* wr = x_proj_w + j * E_DIM;
    float acc = 0.0f;
#pragma unroll 4
    for (int e = 0; e < E_DIM; ++e) acc += xr[e] * wr[e];
    x_dbl[i] = acc;
}

__global__ void dtproj_kernel(const float* __restrict__ x_dbl,
                              const float* __restrict__ dt_proj_w,
                              float* __restrict__ dt) {
    int i = blockIdx.x * blockDim.x + threadIdx.x;
    if (i >= BL * E_DIM) return;
    int e   = i % E_DIM;
    int row = i / E_DIM;
    const float* xr = x_dbl + (size_t)row * XPROJ_OUT;
    const float* wr = dt_proj_w + e * R_DIM;
    float acc = 0.0f;
#pragma unroll
    for (int r = 0; r < R_DIM; ++r) acc += xr[r] * wr[r];
    dt[i] = acc;
}

// ---------------------------------------------------------------------------
// Selective scan. grid = BATCH*4*6 blocks of 32 threads.
// block -> (b, d, channel-chunk); lane owns channel e, 16-state in VGPRs.
// Scatter-accumulate y into original spatial order via atomic f32 add.
// ---------------------------------------------------------------------------

__global__ void scan_kernel(const float* __restrict__ dt,
                            const float* __restrict__ x_conv,
                            const float* __restrict__ x_dbl,
                            const float* __restrict__ A_log,
                            const float* __restrict__ dt_proj_b,
                            const float* __restrict__ dir_Bs,
                            const int*   __restrict__ orders,
                            const int*   __restrict__ dcodes,
                            float* __restrict__ y_sum) {
    const float LOG2E = 1.4426950408889634f;
    int blk   = blockIdx.x;
    int chunk = blk % (E_DIM / 32);   // 0..5
    int bd    = blk / (E_DIM / 32);   // 0..31
    int d     = bd & 3;
    int b     = bd >> 2;
    int e     = chunk * 32 + (threadIdx.x & 31);

    // Per-channel constants: A2[n] = -exp(A_log[e][n]) * log2(e)
    float A2[D_STATE];
#pragma unroll
    for (int n = 0; n < D_STATE; ++n)
        A2[n] = -__expf(A_log[e * D_STATE + n]) * LOG2E;
    const float dtb = dt_proj_b[e];

    float s[D_STATE];
#pragma unroll
    for (int n = 0; n < D_STATE; ++n) s[n] = 0.0f;

    const int*   ord  = orders + d * L_SEQ;
    const int*   dcd  = dcodes + d * L_SEQ;
    const float* dt_b = dt     + (size_t)b * L_SEQ * E_DIM;
    const float* u_b  = x_conv + (size_t)b * L_SEQ * E_DIM;
    const float* xd_b = x_dbl  + (size_t)b * L_SEQ * XPROJ_OUT;
    float*       y_b  = y_sum  + (size_t)b * L_SEQ * E_DIM;

    for (int t = 0; t < L_SEQ; ++t) {
        int idx = ord[t];
        int dc  = dcd[t];

        // prefetch next gathered rows (global_prefetch_b8) to shorten the chain
        int tn  = (t + 4 < L_SEQ) ? t + 4 : (L_SEQ - 1);
        int nidx = ord[tn];
        __builtin_prefetch(dt_b + (size_t)nidx * E_DIM + e, 0, 1);
        __builtin_prefetch(u_b  + (size_t)nidx * E_DIM + e, 0, 1);

        float din   = dt_b[(size_t)idx * E_DIM + e] + dtb;
        float delta = (din > 20.0f) ? din : log1pf(__expf(din));  // softplus
        float u     = u_b[(size_t)idx * E_DIM + e];
        float du    = delta * u;

        const float* bc = xd_b + (size_t)idx * XPROJ_OUT + R_DIM;  // B then C
        const float* db = dir_Bs + dc * D_STATE;

        float y = 0.0f;
#pragma unroll
        for (int n = 0; n < D_STATE; ++n) {
            float Bn = bc[n] + db[n];
            float Cn = bc[D_STATE + n];
            float dA = exp2f(delta * A2[n]);         // v_exp_f32
            s[n] = dA * s[n] + du * Bn;
            y += s[n] * Cn;
        }
        atomicAdd(&y_b[(size_t)idx * E_DIM + e], y); // global_atomic_add_f32
    }
}

// ---------------------------------------------------------------------------
// Gate: yg = (0.25 * sum_d y_d + x_conv * Dp) * silu(z)  -> f16 for out_proj
// ---------------------------------------------------------------------------

__global__ void gate_kernel(const float* __restrict__ y_sum,
                            const float* __restrict__ x_conv,
                            const float* __restrict__ xz,
                            const float* __restrict__ Dp,
                            _Float16* __restrict__ yg) {
    int i = blockIdx.x * blockDim.x + threadIdx.x;
    if (i >= BL * E_DIM) return;
    int e   = i % E_DIM;
    int row = i / E_DIM;
    float z = xz[(size_t)row * (2 * E_DIM) + E_DIM + e];
    float v = 0.25f * y_sum[i] + x_conv[i] * Dp[e];
    float g = z / (1.0f + __expf(-z));
    yg[i] = (_Float16)(v * g);
}

// ---------------------------------------------------------------------------
// Launch
// ---------------------------------------------------------------------------

static inline size_t align256(size_t x) { return (x + 255) & ~(size_t)255; }

extern "C" void kernel_launch(void* const* d_in, const int* in_sizes, int n_in,
                              void* d_out, int out_size, void* d_ws, size_t ws_size,
                              hipStream_t stream) {
    (void)in_sizes; (void)n_in; (void)out_size; (void)ws_size;

    const float* x          = (const float*)d_in[0];
    // d_in[1] = H, d_in[2] = W (compile-time constants here)
    const float* in_proj_w  = (const float*)d_in[3];
    const float* pw1_w      = (const float*)d_in[4];
    const float* pw1_b      = (const float*)d_in[5];
    const float* dw_w       = (const float*)d_in[6];
    const float* pw2_w      = (const float*)d_in[7];
    const float* x_proj_w   = (const float*)d_in[8];
    const float* dt_proj_w  = (const float*)d_in[9];
    const float* dt_proj_b  = (const float*)d_in[10];
    const float* A_log      = (const float*)d_in[11];
    const float* Dp         = (const float*)d_in[12];
    const float* out_proj_w = (const float*)d_in[13];
    const float* dir_Bs     = (const float*)d_in[14];
    const int*   orders     = (const int*)d_in[15];
    const int*   dcodes     = (const int*)d_in[17];

    float* out = (float*)d_out;

    // ---- workspace carve-out -------------------------------------------
    char* base = (char*)d_ws;
    size_t off = 0;
    auto carve = [&](size_t bytes) -> char* {
        char* p = base + off; off += align256(bytes); return p;
    };
    float*    xz       = (float*)   carve((size_t)BL * 2 * E_DIM * 4);   // 50.3 MB
    _Float16* x_h      = (_Float16*)carve((size_t)BL * D_MODEL * 2);     //  6.3 MB
    _Float16* w_in_h   = (_Float16*)carve((size_t)2 * E_DIM * D_MODEL * 2);
    _Float16* w_out_h  = (_Float16*)carve((size_t)D_MODEL * E_DIM * 2);
    float*    h1       = (float*)   carve((size_t)BATCH * MID_DIM * L_SEQ * 4);
    float*    x_conv   = (float*)   carve((size_t)BL * E_DIM * 4);       // 25.2 MB
    float*    x_dbl    = (float*)   carve((size_t)BL * XPROJ_OUT * 4);   //  5.0 MB
    float*    dt       = (float*)   carve((size_t)BL * E_DIM * 4);       // 25.2 MB
    float*    y_sum    = (float*)   carve((size_t)BL * E_DIM * 4);       // 25.2 MB
    _Float16* yg       = (_Float16*)carve((size_t)BL * E_DIM * 2);       // 12.6 MB

    // ---- 1) f16 conversions --------------------------------------------
    {
        int n = BL * D_MODEL;
        f32_to_f16_kernel<<<(n + 255) / 256, 256, 0, stream>>>(x, x_h, n);
        n = 2 * E_DIM * D_MODEL;
        f32_to_f16_kernel<<<(n + 255) / 256, 256, 0, stream>>>(in_proj_w, w_in_h, n);
        n = D_MODEL * E_DIM;
        f32_to_f16_kernel<<<(n + 255) / 256, 256, 0, stream>>>(out_proj_w, w_out_h, n);
    }

    // ---- 2) in_proj GEMM: xz = x @ in_proj_w^T (32768x384, K=96) -------
    {
        // NT=4: block tile 128x64, grid (32768/128, 384/64), LDS 64x96 f16
        dim3 grid(BL / 128, (2 * E_DIM) / 64);
        size_t lds = (size_t)64 * D_MODEL * sizeof(_Float16);   // 12 KB
        wmma_gemm_nt<4><<<grid, 256, lds, stream>>>(
            x_h, w_in_h, xz, BL, 2 * E_DIM, D_MODEL);
    }

    // ---- 3) conv stack --------------------------------------------------
    {
        int n = BATCH * MID_DIM * L_SEQ;
        pw1_kernel<<<(n + 255) / 256, 256, 0, stream>>>(xz, pw1_w, pw1_b, h1);
        dwpw2_kernel<<<BL, E_DIM, 0, stream>>>(h1, dw_w, pw2_w, x_conv);
    }

    // ---- 4) x_dbl + dt projections -------------------------------------
    {
        int n = BL * XPROJ_OUT;
        xdbl_kernel<<<(n + 255) / 256, 256, 0, stream>>>(x_conv, x_proj_w, x_dbl);
        n = BL * E_DIM;
        dtproj_kernel<<<(n + 255) / 256, 256, 0, stream>>>(x_dbl, dt_proj_w, dt);
    }

    // ---- 5) selective scan (zero accumulator first) ---------------------
    {
        int n = BL * E_DIM;
        zero_f32_kernel<<<(n + 255) / 256, 256, 0, stream>>>(y_sum, n);
        scan_kernel<<<BATCH * 4 * (E_DIM / 32), 32, 0, stream>>>(
            dt, x_conv, x_dbl, A_log, dt_proj_b, dir_Bs, orders, dcodes, y_sum);
    }

    // ---- 6) gate + out_proj GEMM: out = yg @ out_proj_w^T (K=192) -------
    {
        int n = BL * E_DIM;
        gate_kernel<<<(n + 255) / 256, 256, 0, stream>>>(y_sum, x_conv, xz, Dp, yg);
        // NT=3: block tile 128x48, grid (32768/128, 96/48), LDS 48x192 f16
        dim3 grid(BL / 128, D_MODEL / 48);
        size_t lds = (size_t)48 * E_DIM * sizeof(_Float16);     // 18 KB
        wmma_gemm_nt<3><<<grid, 256, lds, stream>>>(
            yg, w_out_h, out, BL, D_MODEL, E_DIM);
    }
}
